// Head_27023934226748
// MI455X (gfx1250) — matrix operand order repaired and checked by reading, not compiled, not run on hardware
//
#include <hip/hip_runtime.h>
#include <hip/hip_bf16.h>

typedef __attribute__((ext_vector_type(16))) _Float16 v16h;
typedef __attribute__((ext_vector_type(8)))  float    v8f;

#define B_  4
#define T_  4096
#define C_  1024
#define H_  64
#define SCALE_ 0.03125f   // C^-0.5, folded into Q at projection time
#define NEGBIG_ (-3.0e38f)

union H2U  { unsigned int u; _Float16 h[2]; };
union U4H8 { uint4 u4; _Float16 h[8]; unsigned int u[4]; };

// ---------------------------------------------------------------------------
// Kernel 1: q = (x*Wq)*scale (T,H) f16 ; kT = (x*Wk)^T (H,T) f16 ; v = x*Wv
// grid = B*T/16 blocks, 96 threads (3 waves; wave w -> one of q/k/v)
// ---------------------------------------------------------------------------
__global__ __launch_bounds__(96) void proj_qkv_kernel(
    const float* __restrict__ x,
    const float* __restrict__ Wq, const float* __restrict__ Wk,
    const float* __restrict__ Wv,
    _Float16* __restrict__ qbuf, _Float16* __restrict__ kT,
    _Float16* __restrict__ vbuf)
{
    const int wave = threadIdx.x >> 5;
    const int lane = threadIdx.x & 31;
    const int half = lane >> 4;
    const int l16  = lane & 15;
    const int tile = blockIdx.x;                   // 16-row tile over flattened B*T
    const size_t rowg = (size_t)tile * 16 + l16;   // A-fragment row

    const float* W = (wave == 0) ? Wq : (wave == 1) ? Wk : Wv;
    const float sc = (wave == 0) ? SCALE_ : 1.0f;

    v8f acc[4] = {};
    const int kl = l16 + 16 * half;                // B-fragment k index within 32-chunk

    for (int kk = 0; kk < C_; kk += 32) {
        // A fragment (16x32 f16): lane row = l16, pairs along K contiguous in x
        v16h a;
        #pragma unroll
        for (int j = 0; j < 8; ++j) {
            const int k0 = ((j >> 2) << 4) + (half << 3) + ((j & 3) << 1);
            const float2 f = *reinterpret_cast<const float2*>(&x[rowg * C_ + kk + k0]);
            a[2 * j]     = (_Float16)f.x;
            a[2 * j + 1] = (_Float16)f.y;
        }
        #pragma unroll
        for (int nt = 0; nt < 4; ++nt) {
            // B fragment (32x16): lane n = l16, element pairs along K in table order
            v16h bfr;
            #pragma unroll
            for (int j = 0; j < 8; ++j) {
                const int k0 = ((j >> 2) << 4) + (half << 3) + ((j & 3) << 1);
                bfr[2 * j]     = (_Float16)W[(size_t)(kk + k0) * H_ + nt * 16 + l16];
                bfr[2 * j + 1] = (_Float16)W[(size_t)(kk + k0 + 1) * H_ + nt * 16 + l16];
            }
            acc[nt] = __builtin_amdgcn_wmma_f32_16x16x32_f16(
                false, a, false, bfr, (short)0, acc[nt], false, false);
        }
    }

    // D layout: lane n = l16 ; row m = r + 8*half
    const int b = tile >> 8;                       // 256 tiles per batch
    #pragma unroll
    for (int r = 0; r < 8; ++r) {
        const int m    = r + 8 * half;
        const int trow = tile * 16 + m;            // flat row over B*T
        const int t    = trow & (T_ - 1);
        #pragma unroll
        for (int nt = 0; nt < 4; ++nt) {
            const int n = nt * 16 + l16;
            const _Float16 hv = (_Float16)(acc[nt][r] * sc);
            if (wave == 0)       qbuf[(size_t)trow * H_ + n] = hv;
            else if (wave == 1)  kT[((size_t)b * H_ + n) * T_ + t] = hv;
            else                 vbuf[(size_t)trow * H_ + n] = hv;
        }
    }
}

// ---------------------------------------------------------------------------
// Kernel 2: causal flash attention. 256 blocks x 128 threads; each wave owns
// one 16-query tile and streams 32-key blocks with online softmax.
// Softmax ownership is split: lane l handles keys 0-15 of row l&15,
// lane l+16 handles keys 16-31 (no duplicated exp work).
// ---------------------------------------------------------------------------
__global__ __launch_bounds__(128) void flash_attn_kernel(
    const _Float16* __restrict__ qbuf, const _Float16* __restrict__ kT,
    const _Float16* __restrict__ vbuf, float* __restrict__ out)
{
    __shared__ float    ldsS[4][16][36];   // per-wave S staging (36: distinct banks, 16B rows)
    __shared__ _Float16 ldsP[4][16][40];   // per-wave P staging (40: distinct banks, 16B rows)

    const int wave = threadIdx.x >> 5;
    const int lane = threadIdx.x & 31;
    const int half = lane >> 4;
    const int l16  = lane & 15;

    const int gt = blockIdx.x * 4 + wave;  // global q-tile id
    const int b  = gt >> 8;                // 256 tiles per batch
    const int qt = gt & 255;

    // Q A-fragments for both 32-wide K halves of H=64 (pairs contiguous in qbuf)
    v16h aQ[2];
    #pragma unroll
    for (int kh = 0; kh < 2; ++kh) {
        #pragma unroll
        for (int j = 0; j < 8; ++j) {
            const int k0 = kh * 32 + ((j >> 2) << 4) + (half << 3) + ((j & 3) << 1);
            H2U c;
            c.u = *reinterpret_cast<const unsigned int*>(
                &qbuf[((size_t)b * T_ + qt * 16 + l16) * H_ + k0]);
            aQ[kh][2 * j]     = c.h[0];
            aQ[kh][2 * j + 1] = c.h[1];
        }
    }

    v8f   acc[4] = {};
    float m_run  = NEGBIG_;
    float l_run  = 0.0f;

    const int myrow = l16;                 // row this lane owns
    const int hbase = half * 16;           // this lane's key sub-range within the block
    const int qidx  = qt * 16 + myrow;
    const int kb_end = (qt * 16 + 15) >> 5;

    for (int kb = 0; kb <= kb_end; ++kb) {
        const int key0 = kb * 32;

        // ---- S = Q * K^T  (two n-tiles of keys, chained over two k-halves)
        v8f s0 = {}, s1 = {};
        #pragma unroll
        for (int kh = 0; kh < 2; ++kh) {
            v16h b0, b1;
            #pragma unroll
            for (int j = 0; j < 8; ++j) {
                const int h = kh * 32 + ((j >> 2) << 4) + (half << 3) + ((j & 3) << 1);
                const size_t base = ((size_t)b * H_ + h) * T_ + key0 + l16;
                b0[2 * j] = kT[base];      b0[2 * j + 1] = kT[base + T_];
                b1[2 * j] = kT[base + 16]; b1[2 * j + 1] = kT[base + T_ + 16];
            }
            s0 = __builtin_amdgcn_wmma_f32_16x16x32_f16(false, aQ[kh], false, b0,
                                                        (short)0, s0, false, false);
            s1 = __builtin_amdgcn_wmma_f32_16x16x32_f16(false, aQ[kh], false, b1,
                                                        (short)0, s1, false, false);
        }

        // ---- stage S through LDS: D-layout -> (row, key-half) per lane
        #pragma unroll
        for (int r = 0; r < 8; ++r) {
            const int m = r + 8 * half;
            ldsS[wave][m][l16]      = s0[r];
            ldsS[wave][m][16 + l16] = s1[r];
        }
        asm volatile("s_wait_dscnt 0" ::: "memory");   // wave-local store->load fence

        // ---- online softmax: 16 keys per lane, combine halves via shfl_xor
        float pv[16];
        float bm = NEGBIG_;
        #pragma unroll
        for (int i = 0; i < 4; ++i) {
            const float4 f = *reinterpret_cast<const float4*>(&ldsS[wave][myrow][hbase + 4 * i]);
            const float v0 = (key0 + hbase + 4 * i + 0 <= qidx) ? f.x : NEGBIG_;
            const float v1 = (key0 + hbase + 4 * i + 1 <= qidx) ? f.y : NEGBIG_;
            const float v2 = (key0 + hbase + 4 * i + 2 <= qidx) ? f.z : NEGBIG_;
            const float v3 = (key0 + hbase + 4 * i + 3 <= qidx) ? f.w : NEGBIG_;
            pv[4 * i + 0] = v0; pv[4 * i + 1] = v1;
            pv[4 * i + 2] = v2; pv[4 * i + 3] = v3;
            bm = fmaxf(bm, fmaxf(fmaxf(v0, v1), fmaxf(v2, v3)));
        }
        bm = fmaxf(bm, __shfl_xor(bm, 16, 32));        // combine the two key halves

        const float m_new = fmaxf(m_run, bm);
        const float corr  = __expf(m_run - m_new);
        float ls = 0.0f;
        #pragma unroll
        for (int j = 0; j < 16; ++j) {
            const float e = __expf(pv[j] - m_new);
            pv[j] = e;
            ls += e;
        }
        ls += __shfl_xor(ls, 16, 32);                  // combine the two key halves
        l_run = l_run * corr + ls;
        m_run = m_new;

        // ---- rescale accumulator rows (broadcast corr from row-owner lane)
        #pragma unroll
        for (int r = 0; r < 8; ++r) {
            const int m = r + 8 * half;
            const float cm = __shfl(corr, m, 32);
            #pragma unroll
            for (int nt = 0; nt < 4; ++nt) acc[nt][r] *= cm;
        }

        // ---- stage P (f16) through LDS: owner writes its 16 keys as 2x b128
        unsigned int parr[8];
        #pragma unroll
        for (int j = 0; j < 8; ++j) {
            H2U t;
            t.h[0] = (_Float16)pv[2 * j];
            t.h[1] = (_Float16)pv[2 * j + 1];
            parr[j] = t.u;
        }
        *reinterpret_cast<uint4*>(&ldsP[wave][myrow][hbase]) =
            make_uint4(parr[0], parr[1], parr[2], parr[3]);
        *reinterpret_cast<uint4*>(&ldsP[wave][myrow][hbase + 8]) =
            make_uint4(parr[4], parr[5], parr[6], parr[7]);
        asm volatile("s_wait_dscnt 0" ::: "memory");

        // ---- P A-fragment: two contiguous 16B runs per lane (A-layout exact match)
        U4H8 r0, r1;
        r0.u4 = *reinterpret_cast<const uint4*>(&ldsP[wave][l16][half * 8]);
        r1.u4 = *reinterpret_cast<const uint4*>(&ldsP[wave][l16][16 + half * 8]);
        v16h aP;
        #pragma unroll
        for (int i = 0; i < 8; ++i) {
            aP[i]     = r0.h[i];
            aP[8 + i] = r1.h[i];
        }

        // ---- O += P * V   (B-fragment: lane n = h, element pairs along keys in table order)
        #pragma unroll
        for (int nt = 0; nt < 4; ++nt) {
            v16h bV;
            #pragma unroll
            for (int j = 0; j < 8; ++j) {
                const int k0 = ((j >> 2) << 4) + (half << 3) + ((j & 3) << 1);
                const size_t base = ((size_t)b * T_ + key0 + k0) * H_ + nt * 16 + l16;
                bV[2 * j] = vbuf[base]; bV[2 * j + 1] = vbuf[base + H_];
            }
            acc[nt] = __builtin_amdgcn_wmma_f32_16x16x32_f16(
                false, aP, false, bV, (short)0, acc[nt], false, false);
        }
    }

    // ---- normalize and store fp32 output
    const float invl = 1.0f / l_run;
    #pragma unroll
    for (int r = 0; r < 8; ++r) {
        const int m = r + 8 * half;
        const float im = __shfl(invl, m, 32);
        const size_t trow = (size_t)b * T_ + qt * 16 + m;
        #pragma unroll
        for (int nt = 0; nt < 4; ++nt)
            out[trow * H_ + nt * 16 + l16] = acc[nt][r] * im;
    }
}

// ---------------------------------------------------------------------------
extern "C" void kernel_launch(void* const* d_in, const int* in_sizes, int n_in,
                              void* d_out, int out_size, void* d_ws, size_t ws_size,
                              hipStream_t stream) {
    const float* x  = (const float*)d_in[0];
    const float* Wk = (const float*)d_in[1];
    const float* Wq = (const float*)d_in[2];
    const float* Wv = (const float*)d_in[3];
    float* out = (float*)d_out;

    const size_t qkv_elems = (size_t)B_ * T_ * H_;          // 1,048,576 f16 each
    _Float16* qbuf = (_Float16*)d_ws;
    _Float16* kT   = qbuf + qkv_elems;
    _Float16* vbuf = kT   + qkv_elems;

    proj_qkv_kernel<<<(B_ * T_) / 16, 96, 0, stream>>>(x, Wq, Wk, Wv, qbuf, kT, vbuf);
    flash_attn_kernel<<<(B_ * (T_ / 16)) / 4, 128, 0, stream>>>(qbuf, kT, vbuf, out);
}
